// GIN_84464826843159
// MI455X (gfx1250) — matrix operand order repaired and checked by reading, not compiled, not run on hardware
//
#include <hip/hip_runtime.h>
#include <hip/hip_bf16.h>
#include <stdint.h>

// ---------------------------------------------------------------------------
// GIN on MI455X (gfx1250, wave32, WMMA).
//  - scatter: wave-per-edge float4 gather + global_atomic_add_f32 (HBM bound)
//  - per-layer fused MLP: bf16x3-split WMMA GEMMs (fp32-accurate), fused
//    eps/bias/ReLU/BN/ReLU epilogues, wave-private LDS tiles (no barriers)
//  - pool: wave-per-node atomics; head: same WMMA kernel, NT2=4
// Workspace use: ~103 MB (agg + x ping buffer + split/transposed weights).
// ---------------------------------------------------------------------------

#define HIDF   128
#define OUTF   64
#define LAYERS 4
#define TM     64      // rows per workgroup (4 waves x 16)
#define LSTR   136     // LDS row stride (bf16 elems): 272B rows -> conflict-free b128
#define BN_EPS_F 1e-5f

typedef __bf16 bf16_t;
typedef __attribute__((ext_vector_type(16))) __bf16 v16bf;
typedef __attribute__((ext_vector_type(8)))  __bf16 v8bf;
typedef __attribute__((ext_vector_type(8)))  float  v8f;

__device__ __forceinline__ v8f wmma_bf16_f32(v16bf a, v16bf b, v8f c) {
  // D = A(16x32 bf16) * B(32x16 bf16) + C(16x16 f32)
  return __builtin_amdgcn_wmma_f32_16x16x32_bf16(
      /*neg_a=*/false, a, /*neg_b=*/false, b,
      /*c_mod=*/(short)0, c, /*reuse_a=*/false, /*reuse_b=*/false);
}

// A fragment from a row-major [rows][LSTR] bf16 LDS tile.
// A layout used here (contiguous k, not ISA table order): lane<16 : row=lane,   K=k0..k0+15 (contiguous)
//                            lane>=16: row=lane-16,K=k0+16..k0+31
__device__ __forceinline__ v16bf load_a_frag(const bf16_t* tile, int m0, int k0,
                                             int lane) {
  const int row = m0 + (lane & 15);
  const int s   = (lane >> 4) * 16;
  const bf16_t* p = tile + row * LSTR + k0 + s;   // 16B aligned
  union { v16bf v; v8bf h[2]; } u;
  u.h[0] = *(const v8bf*)(p);        // 8 bf16 = 16B  (ds_load_b128)
  u.h[1] = *(const v8bf*)(p + 8);    // 8 bf16 = 16B
  return u.v;
}

// B fragment from row-major W^T [N][128] bf16 in global memory.
// B layout used here (same contiguous k as A): lane<16 : col=lane,   K=k0..k0+15 (contiguous)
//                            lane>=16: col=lane-16,K=k0+16..k0+31
__device__ __forceinline__ v16bf load_b_frag(const bf16_t* wt, int n0, int k0,
                                             int lane) {
  const bf16_t* p =
      wt + (size_t)(n0 + (lane & 15)) * HIDF + k0 + (lane >> 4) * 16;
  return *(const v16bf*)p;           // 32B contiguous, 32B aligned
}

// ---------------------------------------------------------------------------
// Fused MLP kernel. GIN=true: h=(1+eps)x+agg, BN between the two ReLUs,
// output ReLU, NT2=8 (out width 128). GIN=false (head): NT2=4 (width 64),
// ReLU after GEMM1 only.
// ---------------------------------------------------------------------------
template <bool GIN, int NT2>
__global__ __launch_bounds__(128) void mlp_fused_kernel(
    const float* __restrict__ xin, const float* __restrict__ agg,
    const float* __restrict__ epsp,
    const bf16_t* __restrict__ w1h, const bf16_t* __restrict__ w1l,
    const float* __restrict__ b1,
    const float* __restrict__ bn_gamma, const float* __restrict__ bn_beta,
    const float* __restrict__ bn_mean, const float* __restrict__ bn_var,
    const bf16_t* __restrict__ w2h, const bf16_t* __restrict__ w2l,
    const float* __restrict__ b2,
    float* __restrict__ xout, int nrows) {
  __shared__ bf16_t sh[TM * LSTR];   // hi halves of activations
  __shared__ bf16_t sl[TM * LSTR];   // lo residuals
  const int lane = threadIdx.x & 31;
  const int wave = threadIdx.x >> 5;
  const int m0   = wave * 16;                 // wave-private 16-row slice
  const int m0g  = blockIdx.x * TM;
  const v8f vzero = {0.f, 0.f, 0.f, 0.f, 0.f, 0.f, 0.f, 0.f};

  float e1 = 1.0f;
  if constexpr (GIN) e1 = 1.0f + epsp[0];

  // ---- stage wave's 16 rows of h_pre into LDS as bf16 hi/lo split ----
  for (int rr = 0; rr < 16; ++rr) {
    const int row = m0 + rr;
    int grow = m0g + row;
    if (grow >= nrows) grow = 0;  // clamp; garbage rows never stored
    const float4 xv = *(const float4*)(xin + (size_t)grow * HIDF + lane * 4);
    float f[4] = {xv.x, xv.y, xv.z, xv.w};
    if constexpr (GIN) {
      const float4 av = *(const float4*)(agg + (size_t)grow * HIDF + lane * 4);
      f[0] = e1 * f[0] + av.x; f[1] = e1 * f[1] + av.y;
      f[2] = e1 * f[2] + av.z; f[3] = e1 * f[3] + av.w;
    }
    bf16_t* ph = sh + row * LSTR + lane * 4;
    bf16_t* pl = sl + row * LSTR + lane * 4;
#pragma unroll
    for (int j = 0; j < 4; ++j) {
      const bf16_t hh = (bf16_t)f[j];
      ph[j] = hh;
      pl[j] = (bf16_t)(f[j] - (float)hh);
    }
  }
  // No barrier: each wave reads only rows it wrote; dscnt ordering suffices.

  const int nlo = lane & 15;
  const int ro  = (lane >> 4) * 8;

  // ---- GEMM1: [16x128] @ [128x128], bf16x3 split, fp32 accumulate ----
  v8f acc[8];
#pragma unroll
  for (int nt = 0; nt < 8; ++nt) acc[nt] = vzero;
  for (int ks = 0; ks < 4; ++ks) {
    const v16bf ah = load_a_frag(sh, m0, ks * 32, lane);
    const v16bf al = load_a_frag(sl, m0, ks * 32, lane);
#pragma unroll
    for (int nt = 0; nt < 8; ++nt) {
      const v16bf bh = load_b_frag(w1h, nt * 16, ks * 32, lane);
      const v16bf bl = load_b_frag(w1l, nt * 16, ks * 32, lane);
      acc[nt] = wmma_bf16_f32(ah, bh, acc[nt]);
      acc[nt] = wmma_bf16_f32(ah, bl, acc[nt]);
      acc[nt] = wmma_bf16_f32(al, bh, acc[nt]);
    }
  }

  // ---- epilogue 1: +b1, ReLU, (BN, ReLU), re-split into LDS ----
#pragma unroll
  for (int nt = 0; nt < 8; ++nt) {
    const int n = nt * 16 + nlo;
    const float bias = b1[n];
    float sc = 0.f, sf = 0.f;
    if constexpr (GIN) {
      sc = bn_gamma[n] * rsqrtf(bn_var[n] + BN_EPS_F);
      sf = bn_beta[n] - bn_mean[n] * sc;
    }
#pragma unroll
    for (int r = 0; r < 8; ++r) {
      float c = acc[nt][r] + bias;
      c = fmaxf(c, 0.0f);
      if constexpr (GIN) {
        c = fmaf(c, sc, sf);
        c = fmaxf(c, 0.0f);
      }
      const int row = m0 + ro + r;     // f32 C layout: VGPR r -> M=r(+8)
      const bf16_t hh = (bf16_t)c;
      sh[row * LSTR + n] = hh;
      sl[row * LSTR + n] = (bf16_t)(c - (float)hh);
    }
  }

  // ---- GEMM2: [16x128] @ [128x(NT2*16)] ----
#pragma unroll
  for (int nt = 0; nt < NT2; ++nt) acc[nt] = vzero;
  for (int ks = 0; ks < 4; ++ks) {
    const v16bf ah = load_a_frag(sh, m0, ks * 32, lane);
    const v16bf al = load_a_frag(sl, m0, ks * 32, lane);
#pragma unroll
    for (int nt = 0; nt < NT2; ++nt) {
      const v16bf bh = load_b_frag(w2h, nt * 16, ks * 32, lane);
      const v16bf bl = load_b_frag(w2l, nt * 16, ks * 32, lane);
      acc[nt] = wmma_bf16_f32(ah, bh, acc[nt]);
      acc[nt] = wmma_bf16_f32(ah, bl, acc[nt]);
      acc[nt] = wmma_bf16_f32(al, bh, acc[nt]);
    }
  }

  // ---- epilogue 2: +b2, (ReLU), store fp32 ----
#pragma unroll
  for (int nt = 0; nt < NT2; ++nt) {
    const int n = nt * 16 + nlo;
    const float bias = b2[n];
#pragma unroll
    for (int r = 0; r < 8; ++r) {
      float c = acc[nt][r] + bias;
      if constexpr (GIN) c = fmaxf(c, 0.0f);
      const int grow = m0g + m0 + ro + r;
      if (grow < nrows) xout[(size_t)grow * (NT2 * 16) + n] = c;
    }
  }
}

// ---------------------------------------------------------------------------
// Edge scatter: one wave per edge, lane = float4 of the 128-wide feature.
// ---------------------------------------------------------------------------
__global__ void scatter_kernel(const float* __restrict__ x,
                               const int* __restrict__ ei,
                               float* __restrict__ agg, int nedges) {
  const int gid = blockIdx.x * blockDim.x + threadIdx.x;
  const int e = gid >> 5, lane = gid & 31;
  if (e >= nedges) return;
  const int s = ei[e];
  const int d = ei[nedges + e];
  const float4 v = *(const float4*)(x + (size_t)s * HIDF + lane * 4);
  float* a = agg + (size_t)d * HIDF + lane * 4;
  atomicAdd(a + 0, v.x);
  atomicAdd(a + 1, v.y);
  atomicAdd(a + 2, v.z);
  atomicAdd(a + 3, v.w);
}

// Graph pooling: one wave per node.
__global__ void pool_kernel(const float* __restrict__ x,
                            const int* __restrict__ batch,
                            float* __restrict__ g, int nnodes) {
  const int gid = blockIdx.x * blockDim.x + threadIdx.x;
  const int nidx = gid >> 5, lane = gid & 31;
  if (nidx >= nnodes) return;
  const int b = batch[nidx];
  const float4 v = *(const float4*)(x + (size_t)nidx * HIDF + lane * 4);
  float* a = g + (size_t)b * HIDF + lane * 4;
  atomicAdd(a + 0, v.x);
  atomicAdd(a + 1, v.y);
  atomicAdd(a + 2, v.z);
  atomicAdd(a + 3, v.w);
}

__global__ void zero_kernel(float* __restrict__ p, int n) {
  const int i = blockIdx.x * blockDim.x + threadIdx.x;
  if (i < n) p[i] = 0.0f;
}

// Transpose + bf16 hi/lo split: W[K][N] fp32 -> WT_hi/lo[N][K] bf16.
__global__ void prep_wt_kernel(const float* __restrict__ W,
                               bf16_t* __restrict__ hi,
                               bf16_t* __restrict__ lo, int K, int N) {
  const int idx = blockIdx.x * blockDim.x + threadIdx.x;
  if (idx >= K * N) return;
  const int n = idx / K, k = idx - n * K;
  const float w = W[(size_t)k * N + n];
  const bf16_t h = (bf16_t)w;
  hi[idx] = h;
  lo[idx] = (bf16_t)(w - (float)h);
}

// ---------------------------------------------------------------------------
extern "C" void kernel_launch(void* const* d_in, const int* in_sizes, int n_in,
                              void* d_out, int out_size, void* d_ws,
                              size_t ws_size, hipStream_t stream) {
  (void)n_in; (void)ws_size;
  const float* x     = (const float*)d_in[0];
  const int*   ei    = (const int*)d_in[1];
  const int*   batch = (const int*)d_in[2];
  const float* eps   = (const float*)d_in[3];
  const float* W1    = (const float*)d_in[4];
  const float* b1    = (const float*)d_in[5];
  const float* gamma = (const float*)d_in[6];
  const float* beta  = (const float*)d_in[7];
  const float* bnm   = (const float*)d_in[8];
  const float* bnv   = (const float*)d_in[9];
  const float* W2    = (const float*)d_in[10];
  const float* b2    = (const float*)d_in[11];
  const float* l1w   = (const float*)d_in[12];
  const float* l1b   = (const float*)d_in[13];
  const float* l2w   = (const float*)d_in[14];
  const float* l2b   = (const float*)d_in[15];
  float* out = (float*)d_out;

  const int nnodes  = in_sizes[0] / HIDF;
  const int nedges  = in_sizes[1] / 2;
  const int ngraphs = out_size / OUTF;

  // ---- workspace carve-up (256B aligned) ----
  char* ws = (char*)d_ws;
  size_t off = 0;
  auto carve = [&](size_t bytes) -> void* {
    void* p = ws + off;
    off = (off + bytes + 255) & ~(size_t)255;
    return p;
  };
  float*  agg  = (float*)carve((size_t)nnodes * HIDF * 4);
  float*  xbuf = (float*)carve((size_t)nnodes * HIDF * 4);
  float*  g    = (float*)carve((size_t)ngraphs * HIDF * 4);
  bf16_t* w1h  = (bf16_t*)carve((size_t)LAYERS * HIDF * HIDF * 2);
  bf16_t* w1l  = (bf16_t*)carve((size_t)LAYERS * HIDF * HIDF * 2);
  bf16_t* w2h  = (bf16_t*)carve((size_t)LAYERS * HIDF * HIDF * 2);
  bf16_t* w2l  = (bf16_t*)carve((size_t)LAYERS * HIDF * HIDF * 2);
  bf16_t* l1h  = (bf16_t*)carve((size_t)HIDF * HIDF * 2);
  bf16_t* l1l  = (bf16_t*)carve((size_t)HIDF * HIDF * 2);
  bf16_t* l2h  = (bf16_t*)carve((size_t)OUTF * HIDF * 2);
  bf16_t* l2l  = (bf16_t*)carve((size_t)OUTF * HIDF * 2);

  // ---- weight prep: transpose + bf16 hi/lo split ----
  const int wgrid = (HIDF * HIDF + 255) / 256;
  for (int l = 0; l < LAYERS; ++l) {
    const size_t o = (size_t)l * HIDF * HIDF;
    prep_wt_kernel<<<wgrid, 256, 0, stream>>>(W1 + o, w1h + o, w1l + o, HIDF, HIDF);
    prep_wt_kernel<<<wgrid, 256, 0, stream>>>(W2 + o, w2h + o, w2l + o, HIDF, HIDF);
  }
  prep_wt_kernel<<<wgrid, 256, 0, stream>>>(l1w, l1h, l1l, HIDF, HIDF);
  prep_wt_kernel<<<(HIDF * OUTF + 255) / 256, 256, 0, stream>>>(l2w, l2h, l2l,
                                                                HIDF, OUTF);

  // ---- GIN layers ----
  const int zgrid  = (nnodes * HIDF + 255) / 256;
  const int sgrid  = (nedges * 32 + 255) / 256;
  const int mgrid  = (nnodes + TM - 1) / TM;
  const float* xcur = x;
  for (int l = 0; l < LAYERS; ++l) {
    const size_t o = (size_t)l * HIDF * HIDF;
    zero_kernel<<<zgrid, 256, 0, stream>>>(agg, nnodes * HIDF);
    scatter_kernel<<<sgrid, 256, 0, stream>>>(xcur, ei, agg, nedges);
    mlp_fused_kernel<true, 8><<<mgrid, 128, 0, stream>>>(
        xcur, agg, eps + l, w1h + o, w1l + o, b1 + (size_t)l * HIDF,
        gamma + (size_t)l * HIDF, beta + (size_t)l * HIDF,
        bnm + (size_t)l * HIDF, bnv + (size_t)l * HIDF, w2h + o, w2l + o,
        b2 + (size_t)l * HIDF, xbuf, nnodes);
    xcur = xbuf;  // layers 1..3 update in place (WGs own disjoint rows)
  }

  // ---- pool + head ----
  zero_kernel<<<(ngraphs * HIDF + 255) / 256, 256, 0, stream>>>(g,
                                                               ngraphs * HIDF);
  pool_kernel<<<(nnodes * 32 + 255) / 256, 256, 0, stream>>>(xcur, batch, g,
                                                             nnodes);
  mlp_fused_kernel<false, 4><<<(ngraphs + TM - 1) / TM, 128, 0, stream>>>(
      g, nullptr, nullptr, l1h, l1l, l1b, nullptr, nullptr, nullptr, nullptr,
      l2h, l2l, l2b, out, ngraphs);
}
